// PAMNet_13709535609048
// MI455X (gfx1250) — matrix operand order repaired and checked by reading, not compiled, not run on hardware
//
#include <hip/hip_runtime.h>
#include <hip/hip_bf16.h>

typedef _Float16 half_t;
typedef __attribute__((ext_vector_type(16))) _Float16 v16h;
typedef __attribute__((ext_vector_type(8)))  float    v8f;

#define NNODES 8192
#define NGRAPH 64
#define PPG    128
#define PI_F   3.14159265358979f

__device__ __forceinline__ float silu_f(float s) { return s / (1.0f + __expf(-s)); }

// DimeNet envelope, p=5:  1/x - 21 x^4 + 35 x^5 - 15 x^6
__device__ __forceinline__ float envelope5(float x) {
  float x2 = x * x, x4 = x2 * x2;
  return 1.0f / x + x4 * (-21.0f + x * (35.0f - 15.0f * x));
}

// ---------------------------------------------------------------------------
// Pack a row-major f32 weight [NKS*32, 128] into B-fragment order (f16):
// packed[((kk*8+nt)*32 + lane)*16 + e] = W[(kk*32 + (e>>3)*16 + (lane>>4)*8 + (e&7))*128 + nt*16 + (lane&15)]
// so each lane's 16 halves for one 32x16 WMMA B-tile are contiguous.
// ---------------------------------------------------------------------------
__global__ void pack_w_kernel(const float* __restrict__ W, half_t* __restrict__ Wp, int NKS) {
  int i = blockIdx.x * 256 + threadIdx.x;
  int total = NKS * 8 * 32 * 16;
  if (i >= total) return;
  int e    = i & 15;
  int lane = (i >> 4) & 31;
  int nt   = (i >> 9) & 7;
  int kk   = i >> 12;
  int K = kk * 32 + (e >> 3) * 16 + (lane >> 4) * 8 + (e & 7);
  int n = nt * 16 + (lane & 15);
  Wp[i] = (half_t)W[(size_t)K * 128 + n];
}

__global__ void zero_kernel(float* p, int n) {
  int i = blockIdx.x * 256 + threadIdx.x;
  if (i < n) p[i] = 0.0f;
}

__global__ void copy_kernel(const float* __restrict__ s, float* __restrict__ d, int n) {
  int i = blockIdx.x * 256 + threadIdx.x;
  if (i < n) d[i] = s[i];
}

// ---------------------------------------------------------------------------
// Embedding lookup -> x (f32) and xh (f16)
// ---------------------------------------------------------------------------
__global__ void embed_kernel(const int* __restrict__ xr, const float* __restrict__ emb,
                             float* __restrict__ x, half_t* __restrict__ xh) {
  int i = blockIdx.x * 256 + threadIdx.x;
  if (i >= NNODES * 128) return;
  int n = i >> 7, d = i & 127;
  float v = emb[xr[n] * 128 + d];
  x[i] = v;
  xh[i] = (half_t)v;
}

// ---------------------------------------------------------------------------
// Bessel basis + silu(basis @ W_rbf + b) per edge -> f16 [E,128]; optional dist out
// ---------------------------------------------------------------------------
__global__ void edge_rbf_kernel(const float* __restrict__ pos, const int* __restrict__ src,
                                const int* __restrict__ dst, int E,
                                const float* __restrict__ freq, const float* __restrict__ Wr,
                                const float* __restrict__ br, float cutoff,
                                half_t* __restrict__ rbfh, float* __restrict__ dist_out) {
  int e = blockIdx.x;
  if (e >= E) return;
  __shared__ float basis[16];
  __shared__ float dsh;
  int tid = threadIdx.x;
  if (tid == 0) {
    int i = dst[e], j = src[e];
    float dx = pos[i * 3 + 0] - pos[j * 3 + 0];
    float dy = pos[i * 3 + 1] - pos[j * 3 + 1];
    float dz = pos[i * 3 + 2] - pos[j * 3 + 2];
    float d = sqrtf(fmaxf(dx * dx + dy * dy + dz * dz, 1e-12f));
    dsh = d;
    if (dist_out) dist_out[e] = d;
  }
  __syncthreads();
  if (tid < 16) {
    float xr = dsh / cutoff;
    basis[tid] = envelope5(xr) * __sinf(freq[tid] * xr);
  }
  __syncthreads();
  float acc = br[tid];
#pragma unroll
  for (int k = 0; k < 16; ++k) acc += basis[k] * Wr[k * 128 + tid];
  rbfh[(size_t)e * 128 + tid] = (half_t)silu_f(acc);
}

// ---------------------------------------------------------------------------
// Spherical basis (Legendre x Bessel) per triplet/pair -> f32 [T,48] (42 used)
// angle from (pos[j]-pos[i], pos[k]-pos[j]); radial from dist_l[idx_e]
// ---------------------------------------------------------------------------
__global__ void sph_basis_kernel(const float* __restrict__ pos, const float* __restrict__ dist_l,
                                 const int* __restrict__ ii, const int* __restrict__ jj,
                                 const int* __restrict__ kk_, const int* __restrict__ ee,
                                 int T, float cutoff, float* __restrict__ basis) {
  int t = blockIdx.x * 128 + threadIdx.x;
  if (t >= T) return;
  int i = ii[t], j = jj[t], k = kk_[t];
  float ax = pos[j * 3 + 0] - pos[i * 3 + 0];
  float ay = pos[j * 3 + 1] - pos[i * 3 + 1];
  float az = pos[j * 3 + 2] - pos[i * 3 + 2];
  float bx = pos[k * 3 + 0] - pos[j * 3 + 0];
  float by = pos[k * 3 + 1] - pos[j * 3 + 1];
  float bz = pos[k * 3 + 2] - pos[j * 3 + 2];
  float dt = ax * bx + ay * by + az * bz;
  float cx = ay * bz - az * by;
  float cy = az * bx - ax * bz;
  float cz = ax * by - ay * bx;
  float cn = sqrtf(fmaxf(cx * cx + cy * cy + cz * cz, 1e-12f));
  float ct = dt / sqrtf(dt * dt + cn * cn);  // cos(atan2(cn, dt))
  float P[7];
  P[0] = 1.0f;
  P[1] = ct;
#pragma unroll
  for (int l = 1; l < 6; ++l) P[l + 1] = ((2 * l + 1) * ct * P[l] - l * P[l - 1]) / (float)(l + 1);
  float d = dist_l[ee[t]];
  float xr = d / cutoff;
  float env = envelope5(xr);
  float rb[6];
#pragma unroll
  for (int r = 0; r < 6; ++r) rb[r] = env * __sinf(PI_F * (float)(r + 1) * xr);
  float* bp = basis + (size_t)t * 48;
#pragma unroll
  for (int s = 0; s < 7; ++s)
#pragma unroll
    for (int r = 0; r < 6; ++r) bp[s * 6 + r] = P[s] * rb[r];
}

// ---------------------------------------------------------------------------
// Fused: silu(basis @ W_sbf + b) * m_base[gather] scatter-added into m_tot[scatter]
// ---------------------------------------------------------------------------
__global__ void sph_scatter_kernel(const float* __restrict__ basis, const float* __restrict__ Wsbf,
                                   const float* __restrict__ bsbf, const float* __restrict__ m_base,
                                   float* __restrict__ m_tot, const int* __restrict__ idx_g,
                                   const int* __restrict__ idx_s, int T) {
  int t = blockIdx.x;
  if (t >= T) return;
  __shared__ float bs[48];
  int tid = threadIdx.x;
  if (tid < 42) bs[tid] = basis[(size_t)t * 48 + tid];
  __syncthreads();
  float acc = bsbf[tid];
#pragma unroll
  for (int k = 0; k < 42; ++k) acc += bs[k] * Wsbf[k * 128 + tid];
  float val = silu_f(acc) * m_base[(size_t)idx_g[t] * 128 + tid];
  atomicAdd(&m_tot[(size_t)idx_s[t] * 128 + tid], val);
}

// ---------------------------------------------------------------------------
// Edge-message WMMA GEMM: rows = 16 edges, K = 384 (x[dst] | x[src] | rbf), N = 128.
// If m_out != nullptr: m_out[e,:] = silu(row@W+b); else atomicAdd into agg[dst[e],:].
// 128 threads = 4 waves; wave w covers cols [w*32, w*32+32).
// ---------------------------------------------------------------------------
__global__ void edge_mlp_gemm(const half_t* __restrict__ xh, const half_t* __restrict__ rbfh,
                              const int* __restrict__ src, const int* __restrict__ dst,
                              const half_t* __restrict__ Wp, const float* __restrict__ bias,
                              float* __restrict__ m_out, float* __restrict__ agg, int E) {
  __shared__ half_t As[16][392];
  int tid = threadIdx.x;
  int lane = tid & 31, wave = tid >> 5;
  int e0 = blockIdx.x * 16;

  for (int idx = tid; idx < 16 * 384; idx += 128) {
    int r = idx / 384, c = idx - r * 384;
    int e = e0 + r;
    half_t v = (half_t)0.0f;
    if (e < E) {
      if (c < 128)       v = xh[(size_t)dst[e] * 128 + c];
      else if (c < 256)  v = xh[(size_t)src[e] * 128 + (c - 128)];
      else               v = rbfh[(size_t)e * 128 + (c - 256)];
    }
    As[r][c] = v;
  }
  __syncthreads();

  int g = lane >> 4, mrow = lane & 15;
  v8f c0 = {};
  v8f c1 = {};
#pragma unroll
  for (int kk = 0; kk < 12; ++kk) {
    v16h a;
    int base = kk * 32 + g * 8;
#pragma unroll
    for (int e = 0; e < 8; ++e) a[e] = As[mrow][base + e];
#pragma unroll
    for (int e = 0; e < 8; ++e) a[8 + e] = As[mrow][base + 16 + e];
    const v16h b0 = *(const v16h*)(Wp + ((size_t)((kk * 8 + wave * 2 + 0) * 32 + lane)) * 16);
    const v16h b1 = *(const v16h*)(Wp + ((size_t)((kk * 8 + wave * 2 + 1) * 32 + lane)) * 16);
    if (kk < 11) __builtin_prefetch((const void*)(Wp + (size_t)(kk + 1) * 8 * 32 * 16), 0, 1);
    c0 = __builtin_amdgcn_wmma_f32_16x16x32_f16(false, a, false, b0, (short)0, c0, false, false);
    c1 = __builtin_amdgcn_wmma_f32_16x16x32_f16(false, a, false, b1, (short)0, c1, false, false);
  }

  int n0 = wave * 32 + (lane & 15);
  int n1 = n0 + 16;
  float bb0 = bias[n0], bb1 = bias[n1];
#pragma unroll
  for (int r = 0; r < 8; ++r) {
    int mm = g * 8 + r;
    int e = e0 + mm;
    if (e < E) {
      float v0 = silu_f(c0[r] + bb0);
      float v1 = silu_f(c1[r] + bb1);
      if (m_out) {
        m_out[(size_t)e * 128 + n0] = v0;
        m_out[(size_t)e * 128 + n1] = v1;
      } else {
        int node = dst[e];
        atomicAdd(&agg[(size_t)node * 128 + n0], v0);
        atomicAdd(&agg[(size_t)node * 128 + n1], v1);
      }
    }
  }
}

// ---------------------------------------------------------------------------
// Node-level WMMA GEMM: Y = A[R,128] @ W[128,128] (+bias).
// mode 1: x += silu(Y); xh updated.   mode 0: outp = use_silu ? silu(Y) : Y.
// ---------------------------------------------------------------------------
__global__ void node_gemm(const float* __restrict__ A, const half_t* __restrict__ Wp,
                          const float* __restrict__ bias, float* __restrict__ x,
                          half_t* __restrict__ xh, float* __restrict__ outp,
                          int R, int mode, int use_silu) {
  __shared__ half_t As[16][136];
  int tid = threadIdx.x;
  int lane = tid & 31, wave = tid >> 5;
  int r0 = blockIdx.x * 16;

  for (int idx = tid; idx < 16 * 128; idx += 128) {
    int r = idx >> 7, c = idx & 127;
    int row = r0 + r;
    As[r][c] = (row < R) ? (half_t)A[(size_t)row * 128 + c] : (half_t)0.0f;
  }
  __syncthreads();

  int g = lane >> 4, mrow = lane & 15;
  v8f c0 = {};
  v8f c1 = {};
#pragma unroll
  for (int kk = 0; kk < 4; ++kk) {
    v16h a;
    int base = kk * 32 + g * 8;
#pragma unroll
    for (int e = 0; e < 8; ++e) a[e] = As[mrow][base + e];
#pragma unroll
    for (int e = 0; e < 8; ++e) a[8 + e] = As[mrow][base + 16 + e];
    const v16h b0 = *(const v16h*)(Wp + ((size_t)((kk * 8 + wave * 2 + 0) * 32 + lane)) * 16);
    const v16h b1 = *(const v16h*)(Wp + ((size_t)((kk * 8 + wave * 2 + 1) * 32 + lane)) * 16);
    c0 = __builtin_amdgcn_wmma_f32_16x16x32_f16(false, a, false, b0, (short)0, c0, false, false);
    c1 = __builtin_amdgcn_wmma_f32_16x16x32_f16(false, a, false, b1, (short)0, c1, false, false);
  }

  int n0 = wave * 32 + (lane & 15);
  int n1 = n0 + 16;
  float bb0 = bias ? bias[n0] : 0.0f;
  float bb1 = bias ? bias[n1] : 0.0f;
#pragma unroll
  for (int r = 0; r < 8; ++r) {
    int row = r0 + g * 8 + r;
    if (row < R) {
      float v0 = c0[r] + bb0;
      float v1 = c1[r] + bb1;
      if (use_silu) { v0 = silu_f(v0); v1 = silu_f(v1); }
      if (mode == 1) {
        float x0 = x[(size_t)row * 128 + n0] + v0;
        float x1 = x[(size_t)row * 128 + n1] + v1;
        x[(size_t)row * 128 + n0] = x0;
        x[(size_t)row * 128 + n1] = x1;
        xh[(size_t)row * 128 + n0] = (half_t)x0;
        xh[(size_t)row * 128 + n1] = (half_t)x1;
      } else {
        outp[(size_t)row * 128 + n0] = v0;
        outp[(size_t)row * 128 + n1] = v1;
      }
    }
  }
}

// ---------------------------------------------------------------------------
// agg[dst[e],:] += m[e,:]
// ---------------------------------------------------------------------------
__global__ void edge_agg_kernel(const float* __restrict__ m, const int* __restrict__ dst,
                                float* __restrict__ agg, int E) {
  int i = blockIdx.x * 256 + threadIdx.x;
  if (i >= E * 128) return;
  int e = i >> 7, d = i & 127;
  atomicAdd(&agg[(size_t)dst[e] * 128 + d], m[i]);
}

// att[n] = x[n,:] . Wa + b
__global__ void att_kernel(const float* __restrict__ x, const float* __restrict__ Wa,
                           const float* __restrict__ ba, float* __restrict__ out, int N) {
  __shared__ float was[128];
  int tid = threadIdx.x;
  was[tid] = Wa[tid];
  __syncthreads();
  int n = blockIdx.x * 128 + tid;
  if (n >= N) return;
  float acc = ba[0];
#pragma unroll 8
  for (int k = 0; k < 128; ++k) acc += x[(size_t)n * 128 + k] * was[k];
  out[n] = acc;
}

// leaky-relu + 2-way softmax fusion:  last = og*w_g + ol*w_l
__global__ void combine_kernel(const float* __restrict__ og, const float* __restrict__ ol,
                               const float* __restrict__ ag, const float* __restrict__ al,
                               float* __restrict__ last) {
  int i = blockIdx.x * 256 + threadIdx.x;
  if (i >= NNODES * 128) return;
  int n = i >> 7;
  float zg = ag[n], zl = al[n];
  zg = (zg < 0.0f) ? 0.2f * zg : zg;
  zl = (zl < 0.0f) ? 0.2f * zl : zl;
  float m = fmaxf(zg, zl);
  float eg = __expf(zg - m), el = __expf(zl - m);
  float inv = 1.0f / (eg + el);
  last[i] = og[i] * (eg * inv) + ol[i] * (el * inv);
}

// q = seed @ Wq
__global__ void qv_kernel(const float* __restrict__ seed, const float* __restrict__ Wq,
                          float* __restrict__ qv) {
  int c = threadIdx.x;
  float acc = 0.0f;
#pragma unroll 8
  for (int d = 0; d < 128; ++d) acc += seed[d] * Wq[d * 128 + c];
  qv[c] = acc;
}

// PMA pooling: per-graph 4-head attention over 128 points
__global__ void pma_kernel(const float* __restrict__ Kf, const float* __restrict__ Vf,
                           const float* __restrict__ qv, float* __restrict__ pooled) {
  __shared__ float sc[4][128];
  int b = blockIdx.x, tid = threadIdx.x;
  const float scale = 0.17677669529663687f;  // 1/sqrt(32)
#pragma unroll
  for (int h = 0; h < 4; ++h) {
    float acc = 0.0f;
    const float* kp = Kf + ((size_t)(b * 128 + tid)) * 128 + h * 32;
#pragma unroll
    for (int d = 0; d < 32; ++d) acc += qv[h * 32 + d] * kp[d];
    sc[h][tid] = acc * scale;
  }
  __syncthreads();
  if (tid < 4) {
    float mx = -1e30f;
    for (int p = 0; p < 128; ++p) mx = fmaxf(mx, sc[tid][p]);
    float s = 0.0f;
    for (int p = 0; p < 128; ++p) { float e = __expf(sc[tid][p] - mx); sc[tid][p] = e; s += e; }
    float inv = 1.0f / s;
    for (int p = 0; p < 128; ++p) sc[tid][p] *= inv;
  }
  __syncthreads();
  int h = tid >> 5;
  float acc = 0.0f;
  for (int p = 0; p < 128; ++p) acc += sc[h][p] * Vf[((size_t)(b * 128 + p)) * 128 + tid];
  pooled[b * 128 + tid] = acc;
}

// out = pooled @ Wo
__global__ void final_kernel(const float* __restrict__ pooled, const float* __restrict__ Wo,
                             float* __restrict__ out) {
  __shared__ float ps[128];
  int b = blockIdx.x, n = threadIdx.x;
  ps[n] = pooled[b * 128 + n];
  __syncthreads();
  float acc = 0.0f;
#pragma unroll 8
  for (int d = 0; d < 128; ++d) acc += ps[d] * Wo[d * 128 + n];
  out[b * 128 + n] = acc;
}

// ===========================================================================
extern "C" void kernel_launch(void* const* d_in, const int* in_sizes, int n_in,
                              void* d_out, int out_size, void* d_ws, size_t ws_size,
                              hipStream_t stream) {
  (void)n_in; (void)out_size; (void)ws_size;
  const float* pos     = (const float*)d_in[0];
  const float* emb     = (const float*)d_in[1];
  const float* freq_g  = (const float*)d_in[2];
  const float* freq_l  = (const float*)d_in[3];
  const float* W_rbf_g = (const float*)d_in[4];
  const float* b_rbf_g = (const float*)d_in[5];
  const float* W_rbf_l = (const float*)d_in[6];
  const float* b_rbf_l = (const float*)d_in[7];
  const float* W_sbf1  = (const float*)d_in[8];
  const float* b_sbf1  = (const float*)d_in[9];
  const float* W_sbf2  = (const float*)d_in[10];
  const float* b_sbf2  = (const float*)d_in[11];
  const float* gW_m = (const float*)d_in[12];
  const float* gb_m = (const float*)d_in[13];
  const float* gW_h = (const float*)d_in[14];
  const float* gb_h = (const float*)d_in[15];
  const float* gW_o = (const float*)d_in[16];
  const float* gb_o = (const float*)d_in[17];
  const float* gW_a = (const float*)d_in[18];
  const float* gb_a = (const float*)d_in[19];
  const float* lW_m = (const float*)d_in[20];
  const float* lb_m = (const float*)d_in[21];
  const float* lW_h = (const float*)d_in[22];
  const float* lb_h = (const float*)d_in[23];
  const float* lW_o = (const float*)d_in[24];
  const float* lb_o = (const float*)d_in[25];
  const float* lW_a = (const float*)d_in[26];
  const float* lb_a = (const float*)d_in[27];
  const float* Wq   = (const float*)d_in[28];
  const float* Wk   = (const float*)d_in[29];
  const float* Wv   = (const float*)d_in[30];
  const float* Wo   = (const float*)d_in[31];
  const float* seed = (const float*)d_in[32];
  const int* x_raw  = (const int*)d_in[33];
  const int* eg     = (const int*)d_in[35];
  const int* el     = (const int*)d_in[36];
  const int* idx_i  = (const int*)d_in[37];
  const int* idx_j  = (const int*)d_in[38];
  const int* idx_k  = (const int*)d_in[39];
  const int* idx_kj = (const int*)d_in[40];
  const int* idx_ji = (const int*)d_in[41];
  const int* idx_ip  = (const int*)d_in[42];
  const int* idx_j1p = (const int*)d_in[43];
  const int* idx_j2p = (const int*)d_in[44];
  const int* idx_jjp = (const int*)d_in[45];
  const int* idx_jip = (const int*)d_in[46];

  const int Eg = in_sizes[35] / 2;
  const int El = in_sizes[36] / 2;
  const int T  = in_sizes[37];
  const int Tp = in_sizes[42];
  const int* src_g = eg;        const int* dst_g = eg + Eg;
  const int* src_l = el;        const int* dst_l = el + El;

  // ---- workspace bump allocator ----
  char* wsp = (char*)d_ws;
  auto alloc = [&](size_t bytes) -> void* {
    void* p = (void*)wsp;
    wsp += (bytes + 255) & ~(size_t)255;
    return p;
  };
  float*  x      = (float*)alloc((size_t)NNODES * 128 * 4);
  half_t* xh     = (half_t*)alloc((size_t)NNODES * 128 * 2);
  half_t* rbfg   = (half_t*)alloc((size_t)Eg * 128 * 2);
  half_t* rbfl   = (half_t*)alloc((size_t)El * 128 * 2);
  float*  dist_l = (float*)alloc((size_t)El * 4);
  float*  tb     = (float*)alloc((size_t)T * 48 * 4);
  float*  pb     = (float*)alloc((size_t)Tp * 48 * 4);
  float*  m_base = (float*)alloc((size_t)El * 128 * 4);
  float*  m_tot  = (float*)alloc((size_t)El * 128 * 4);
  float*  agg    = (float*)alloc((size_t)NNODES * 128 * 4);
  float*  og     = (float*)alloc((size_t)NNODES * 128 * 4);
  float*  olb    = (float*)alloc((size_t)NNODES * 128 * 4);
  float*  last   = (float*)alloc((size_t)NNODES * 128 * 4);
  float*  Kf     = (float*)alloc((size_t)NNODES * 128 * 4);
  float*  Vf     = (float*)alloc((size_t)NNODES * 128 * 4);
  float*  ag     = (float*)alloc((size_t)NNODES * 4);
  float*  al     = (float*)alloc((size_t)NNODES * 4);
  float*  qv     = (float*)alloc(128 * 4);
  float*  pooled = (float*)alloc((size_t)NGRAPH * 128 * 4);
  half_t* packM  = (half_t*)alloc((size_t)12 * 8 * 32 * 16 * 2);
  half_t* packH  = (half_t*)alloc((size_t)4 * 8 * 32 * 16 * 2);

  const int NPIX = NNODES * 128;
  const int gN   = (NPIX + 255) / 256;

  // 1) embeddings
  embed_kernel<<<gN, 256, 0, stream>>>(x_raw, emb, x, xh);

  // 2) edge radial features (global + local)
  if (Eg > 0) edge_rbf_kernel<<<Eg, 128, 0, stream>>>(pos, src_g, dst_g, Eg, freq_g, W_rbf_g,
                                                      b_rbf_g, 3.5f, rbfg, nullptr);
  if (El > 0) edge_rbf_kernel<<<El, 128, 0, stream>>>(pos, src_l, dst_l, El, freq_l, W_rbf_l,
                                                      b_rbf_l, 2.2f, rbfl, dist_l);

  // 3) spherical bases (computed once, reused for all 3 layers)
  if (T > 0)  sph_basis_kernel<<<(T + 127) / 128, 128, 0, stream>>>(pos, dist_l, idx_i, idx_j,
                                                                    idx_k, idx_kj, T, 2.2f, tb);
  if (Tp > 0) sph_basis_kernel<<<(Tp + 127) / 128, 128, 0, stream>>>(pos, dist_l, idx_ip, idx_j1p,
                                                                     idx_j2p, idx_jjp, Tp, 2.2f, pb);

  const int packM_n = 12 * 8 * 32 * 16;
  const int packH_n = 4 * 8 * 32 * 16;

  for (int l = 0; l < 3; ++l) {
    // ---- global message passing ----
    pack_w_kernel<<<(packM_n + 255) / 256, 256, 0, stream>>>(gW_m + (size_t)l * 384 * 128, packM, 12);
    zero_kernel<<<gN, 256, 0, stream>>>(agg, NPIX);
    if (Eg > 0)
      edge_mlp_gemm<<<(Eg + 15) / 16, 128, 0, stream>>>(xh, rbfg, src_g, dst_g, packM,
                                                        gb_m + l * 128, nullptr, agg, Eg);
    pack_w_kernel<<<(packH_n + 255) / 256, 256, 0, stream>>>(gW_h + (size_t)l * 128 * 128, packH, 4);
    node_gemm<<<NNODES / 16, 128, 0, stream>>>(agg, packH, gb_h + l * 128, x, xh, nullptr,
                                               NNODES, 1, 1);
    if (l == 2) {
      pack_w_kernel<<<(packH_n + 255) / 256, 256, 0, stream>>>(gW_o + (size_t)2 * 128 * 128, packH, 4);
      node_gemm<<<NNODES / 16, 128, 0, stream>>>(x, packH, gb_o + 2 * 128, nullptr, nullptr, og,
                                                 NNODES, 0, 1);
      att_kernel<<<NNODES / 128, 128, 0, stream>>>(x, gW_a + 2 * 128, gb_a + 2, ag, NNODES);
    }

    // ---- local message passing ----
    pack_w_kernel<<<(packM_n + 255) / 256, 256, 0, stream>>>(lW_m + (size_t)l * 384 * 128, packM, 12);
    if (El > 0) {
      edge_mlp_gemm<<<(El + 15) / 16, 128, 0, stream>>>(xh, rbfl, src_l, dst_l, packM,
                                                        lb_m + l * 128, m_base, nullptr, El);
      copy_kernel<<<(El * 128 + 255) / 256, 256, 0, stream>>>(m_base, m_tot, El * 128);
      if (T > 0)
        sph_scatter_kernel<<<T, 128, 0, stream>>>(tb, W_sbf2, b_sbf2, m_base, m_tot, idx_kj,
                                                  idx_ji, T);
      if (Tp > 0)
        sph_scatter_kernel<<<Tp, 128, 0, stream>>>(pb, W_sbf1, b_sbf1, m_base, m_tot, idx_jjp,
                                                   idx_jip, Tp);
    }
    zero_kernel<<<gN, 256, 0, stream>>>(agg, NPIX);
    if (El > 0)
      edge_agg_kernel<<<(El * 128 + 255) / 256, 256, 0, stream>>>(m_tot, dst_l, agg, El);
    pack_w_kernel<<<(packH_n + 255) / 256, 256, 0, stream>>>(lW_h + (size_t)l * 128 * 128, packH, 4);
    node_gemm<<<NNODES / 16, 128, 0, stream>>>(agg, packH, lb_h + l * 128, x, xh, nullptr,
                                               NNODES, 1, 1);
    if (l == 2) {
      pack_w_kernel<<<(packH_n + 255) / 256, 256, 0, stream>>>(lW_o + (size_t)2 * 128 * 128, packH, 4);
      node_gemm<<<NNODES / 16, 128, 0, stream>>>(x, packH, lb_o + 2 * 128, nullptr, nullptr, olb,
                                                 NNODES, 0, 1);
      att_kernel<<<NNODES / 128, 128, 0, stream>>>(x, lW_a + 2 * 128, lb_a + 2, al, NNODES);
    }
  }

  // 4) attention fusion across paths -> last
  combine_kernel<<<gN, 256, 0, stream>>>(og, olb, ag, al, last);

  // 5) PMA pooling
  pack_w_kernel<<<(packH_n + 255) / 256, 256, 0, stream>>>(Wk, packH, 4);
  node_gemm<<<NNODES / 16, 128, 0, stream>>>(last, packH, nullptr, nullptr, nullptr, Kf,
                                             NNODES, 0, 0);
  pack_w_kernel<<<(packH_n + 255) / 256, 256, 0, stream>>>(Wv, packH, 4);
  node_gemm<<<NNODES / 16, 128, 0, stream>>>(last, packH, nullptr, nullptr, nullptr, Vf,
                                             NNODES, 0, 0);
  qv_kernel<<<1, 128, 0, stream>>>(seed, Wq, qv);
  pma_kernel<<<NGRAPH, 128, 0, stream>>>(Kf, Vf, qv, pooled);
  final_kernel<<<NGRAPH, 128, 0, stream>>>(pooled, Wo, (float*)d_out);
}
